// WeightlessGQA_48241072668864
// MI455X (gfx1250) — matrix operand order repaired and checked by reading, not compiled, not run on hardware
//
#include <hip/hip_runtime.h>
#include <hip/hip_bf16.h>
#include <stdint.h>

// ---------------------------------------------------------------------------
// CDNA5 (gfx1250) wave32 WMMA types
// ---------------------------------------------------------------------------
typedef __bf16 v16bf __attribute__((ext_vector_type(16)));
typedef float  v8f   __attribute__((ext_vector_type(8)));

union FragU {
  v16bf v;
  uint4 q[2];
};

__device__ __forceinline__ unsigned short f32_to_bf16_bits(float f) {
  union { float f; unsigned int u; } c; c.f = f;
  unsigned int u = c.u;
  unsigned int r = u + 0x7FFFu + ((u >> 16) & 1u);  // round-to-nearest-even
  return (unsigned short)(r >> 16);
}

__device__ __forceinline__ v8f wmma_bf16(v16bf a, v16bf b, v8f c) {
  return __builtin_amdgcn_wmma_f32_16x16x32_bf16(
      false, a, false, b, (short)0, c, false, false);
}

// ---------------------------------------------------------------------------
// fp32 -> bf16 conversion (grid-stride)
// ---------------------------------------------------------------------------
__global__ void f32_to_bf16_kernel(const float* __restrict__ src,
                                   unsigned short* __restrict__ dst, size_t n) {
  size_t i = (size_t)blockIdx.x * blockDim.x + threadIdx.x;
  size_t stride = (size_t)gridDim.x * blockDim.x;
  for (; i < n; i += stride) dst[i] = f32_to_bf16_bits(src[i]);
}

// ---------------------------------------------------------------------------
// GEMM: C[M,N] (f32) = A[M,K] (bf16) x W[N,K]^T (bf16)
// block = 256 threads = 8 waves; wave tile 64x64; block tile 256(M) x 128(N)
// Fragments loaded straight from global: operands are L2-resident on MI455X
// (192 MB L2 >> working set), so WMMA density dominates.
// ---------------------------------------------------------------------------
__global__ __launch_bounds__(256) void gemm_bf16_nt_kernel(
    const unsigned short* __restrict__ A,   // [M,K] bf16 bits
    const unsigned short* __restrict__ W,   // [N,K] bf16 bits
    float* __restrict__ C,                  // [M,N] f32
    int M, int N, int K) {
  const int lane  = threadIdx.x & 31;
  const int wid   = threadIdx.x >> 5;
  const int wm    = wid >> 1;              // 0..3
  const int wn    = wid & 1;               // 0..1
  const int m0    = blockIdx.y * 256 + wm * 64;
  const int n0    = blockIdx.x * 128 + wn * 64;

  const int mlane = lane & 15;
  const int aoff  = (lane >> 4) << 4;      // A K-half select: 0 or 16
  const int boff  = (lane >> 4) << 4;      // B K-half select: 0 or 16

  v8f zero = {};
  v8f acc[4][4];
#pragma unroll
  for (int i = 0; i < 4; ++i)
#pragma unroll
    for (int j = 0; j < 4; ++j) acc[i][j] = zero;

  for (int kk = 0; kk < K; kk += 32) {
    FragU af[4], bfr[4];
#pragma unroll
    for (int i = 0; i < 4; ++i) {
      const unsigned short* p = A + (size_t)(m0 + 16 * i + mlane) * K + kk + aoff;
      af[i].q[0] = *(const uint4*)(p);
      af[i].q[1] = *(const uint4*)(p + 8);
    }
#pragma unroll
    for (int j = 0; j < 4; ++j) {
      const unsigned short* p = W + (size_t)(n0 + 16 * j + mlane) * K + kk + boff;
      bfr[j].q[0] = *(const uint4*)(p);
      bfr[j].q[1] = *(const uint4*)(p + 8);
    }
    // speculative prefetch of next K-tile (emits global_prefetch_b8)
    __builtin_prefetch(A + (size_t)(m0 + mlane) * K + kk + 32 + aoff, 0, 1);
    __builtin_prefetch(W + (size_t)(n0 + mlane) * K + kk + 32 + boff, 0, 1);
#pragma unroll
    for (int i = 0; i < 4; ++i)
#pragma unroll
      for (int j = 0; j < 4; ++j)
        acc[i][j] = wmma_bf16(af[i].v, bfr[j].v, acc[i][j]);
  }

  const int ncol = lane & 15;
  const int hi   = lane >> 4;
#pragma unroll
  for (int i = 0; i < 4; ++i)
#pragma unroll
    for (int j = 0; j < 4; ++j)
#pragma unroll
      for (int r = 0; r < 8; ++r)
        C[(size_t)(m0 + 16 * i + r + 8 * hi) * N + (n0 + 16 * j + ncol)] =
            acc[i][j][r];
}

// ---------------------------------------------------------------------------
// RoPE + repack: src f32 [bs*seq, H*128] ->
//   mode 0 (rope): dst bf16 [b,H,seq,128]
//   mode 1 (copy): dst bf16 [b,H,128,seq]   (transposed for PV B-fragments)
// freqs layout: [2, 2048, 64] (cos, sin)
// ---------------------------------------------------------------------------
__global__ void rope_pack_kernel(const float* __restrict__ src,
                                 const float* __restrict__ freqs,
                                 unsigned short* __restrict__ dst,
                                 int H, int mode) {
  const int SEQ = 2048;
  size_t total = (size_t)2 * SEQ * H * 64;
  size_t idx = (size_t)blockIdx.x * blockDim.x + threadIdx.x;
  size_t stride = (size_t)gridDim.x * blockDim.x;
  for (; idx < total; idx += stride) {
    int j = (int)(idx & 63);
    size_t t = idx >> 6;
    int h = (int)(t % H); t /= H;
    int s = (int)(t % SEQ);
    int b = (int)(t / SEQ);
    size_t srow = ((size_t)b * SEQ + s) * ((size_t)H * 128) + (size_t)h * 128 + 2 * j;
    float re = src[srow];
    float im = src[srow + 1];
    if (mode == 0) {
      float c  = freqs[(size_t)s * 64 + j];
      float sn = freqs[(size_t)SEQ * 64 + (size_t)s * 64 + j];
      float ore = re * c - im * sn;
      float oim = re * sn + im * c;
      size_t d = (((size_t)(b * H + h) * SEQ + s) * 128) + 2 * j;
      dst[d]     = f32_to_bf16_bits(ore);
      dst[d + 1] = f32_to_bf16_bits(oim);
    } else {
      size_t d = ((size_t)(b * H + h) * 128 + 2 * j) * SEQ + s;
      dst[d]       = f32_to_bf16_bits(re);
      dst[d + SEQ] = f32_to_bf16_bits(im);
    }
  }
}

// ---------------------------------------------------------------------------
// Flash attention (causal, start_pos = 0), GQA n_rep = 4.
// grid = (seq/128, bs*32); block = 256 (8 waves); each wave owns 16 q rows.
// QK^T and PV via v_wmma_f32_16x16x32_bf16. P transposed to A-fragment layout
// through a wave-private LDS tile (DS ops are in-order within a wave).
// ---------------------------------------------------------------------------
__global__ __launch_bounds__(256) void attn_kernel(
    const unsigned short* __restrict__ qb,    // [2,32,2048,128] bf16
    const unsigned short* __restrict__ kbuf,  // [2, 8,2048,128] bf16
    const unsigned short* __restrict__ vbt,   // [2, 8, 128,2048] bf16
    unsigned short* __restrict__ ob) {        // [2*2048, 4096] bf16
  const int SEQ = 2048, HD = 128, NH = 32, NKV = 8;
  const int lane = threadIdx.x & 31;
  const int wid  = threadIdx.x >> 5;
  const int bh = blockIdx.y;
  const int b  = bh >> 5;
  const int h  = bh & 31;
  const int hkv = h >> 2;
  const int q0 = blockIdx.x * 128 + wid * 16;

  const unsigned short* Q  = qb   + (size_t)(b * NH  + h)   * SEQ * HD;
  const unsigned short* Kp = kbuf + (size_t)(b * NKV + hkv) * SEQ * HD;
  const unsigned short* Vt = vbt  + (size_t)(b * NKV + hkv) * HD  * SEQ;

  __shared__ unsigned short sP[8][16][32];   // per-wave P tile (bf16)

  const int mlane = lane & 15;
  const int aoff  = (lane >> 4) << 4;
  const int boff  = (lane >> 4) << 4;
  const int hi    = lane >> 4;
  const int ncol  = lane & 15;

  // Q A-fragments for all 4 K-steps over hd=128 (held in registers)
  FragU qf[4];
#pragma unroll
  for (int kk = 0; kk < 4; ++kk) {
    const unsigned short* p = Q + (size_t)(q0 + mlane) * HD + kk * 32 + aoff;
    qf[kk].q[0] = *(const uint4*)(p);
    qf[kk].q[1] = *(const uint4*)(p + 8);
  }

  v8f zero = {};
  v8f o[8];
#pragma unroll
  for (int t = 0; t < 8; ++t) o[t] = zero;
  float mi[8], li[8];
#pragma unroll
  for (int r = 0; r < 8; ++r) { mi[r] = -1e30f; li[r] = 0.0f; }

  const float scale = 0.08838834764831845f;  // 1/sqrt(128)

  for (int kb = 0; kb < q0 + 16; kb += 32) {
    // ---- S = Q K^T for 32 keys (two 16x16 tiles) ----
    v8f sa = zero, sb = zero;
#pragma unroll
    for (int kk = 0; kk < 4; ++kk) {
      FragU ka, kb2;
      const unsigned short* pa = Kp + (size_t)(kb + mlane) * HD + kk * 32 + boff;
      ka.q[0] = *(const uint4*)(pa);
      ka.q[1] = *(const uint4*)(pa + 8);
      const unsigned short* pb = Kp + (size_t)(kb + 16 + mlane) * HD + kk * 32 + boff;
      kb2.q[0] = *(const uint4*)(pb);
      kb2.q[1] = *(const uint4*)(pb + 8);
      sa = wmma_bf16(qf[kk].v, ka.v, sa);
      sb = wmma_bf16(qf[kk].v, kb2.v, sb);
    }

    // ---- online softmax (row stats via shfl_xor within 16-lane halves) ----
#pragma unroll
    for (int r = 0; r < 8; ++r) {
      const int qrow = q0 + r + 8 * hi;
      float va = (kb + ncol      <= qrow) ? (sa[r] * scale) : -1e30f;
      float vb = (kb + 16 + ncol <= qrow) ? (sb[r] * scale) : -1e30f;
      float mx = fmaxf(va, vb);
#pragma unroll
      for (int d2 = 1; d2 < 16; d2 <<= 1) mx = fmaxf(mx, __shfl_xor(mx, d2, 32));
      float mnew = fmaxf(mi[r], mx);
      float corr = __expf(mi[r] - mnew);
      float pa = __expf(va - mnew);
      float pb = __expf(vb - mnew);
      float rs = pa + pb;
#pragma unroll
      for (int d2 = 1; d2 < 16; d2 <<= 1) rs += __shfl_xor(rs, d2, 32);
      li[r] = li[r] * corr + rs;
      mi[r] = mnew;
#pragma unroll
      for (int t = 0; t < 8; ++t) o[t][r] *= corr;
      sP[wid][r + 8 * hi][ncol]      = f32_to_bf16_bits(pa);
      sP[wid][r + 8 * hi][ncol + 16] = f32_to_bf16_bits(pb);
    }

    // ---- P (A-fragment via LDS transpose) x V^T (contiguous B-fragments) ----
    FragU pf;
    pf.q[0] = *(const uint4*)&sP[wid][mlane][aoff];
    pf.q[1] = *(const uint4*)&sP[wid][mlane][aoff + 8];
#pragma unroll
    for (int t = 0; t < 8; ++t) {
      FragU vf;
      const unsigned short* pv = Vt + (size_t)(t * 16 + mlane) * SEQ + kb + boff;
      vf.q[0] = *(const uint4*)(pv);
      vf.q[1] = *(const uint4*)(pv + 8);
      o[t] = wmma_bf16(pf.v, vf.v, o[t]);
    }
  }

  // ---- normalize and store to [b*s, h*128 + d] (bf16) ----
#pragma unroll
  for (int r = 0; r < 8; ++r) {
    float inv = 1.0f / li[r];
    const int qrow = q0 + r + 8 * hi;
    size_t base = ((size_t)b * SEQ + qrow) * ((size_t)NH * HD) + (size_t)h * HD;
#pragma unroll
    for (int t = 0; t < 8; ++t)
      ob[base + t * 16 + ncol] = f32_to_bf16_bits(o[t][r] * inv);
  }
}

// ---------------------------------------------------------------------------
// kernel_launch
// ---------------------------------------------------------------------------
extern "C" void kernel_launch(void* const* d_in, const int* in_sizes, int n_in,
                              void* d_out, int out_size, void* d_ws, size_t ws_size,
                              hipStream_t stream) {
  (void)in_sizes; (void)n_in; (void)out_size; (void)ws_size;
  const float* x     = (const float*)d_in[0];
  const float* wq    = (const float*)d_in[1];
  const float* wk    = (const float*)d_in[2];
  const float* wv    = (const float*)d_in[3];
  const float* wo    = (const float*)d_in[4];
  const float* freqs = (const float*)d_in[7];   // [2,2048,64]
  float* out = (float*)d_out;

  const int BS = 2, SEQ = 2048, D = 4096, NH = 32, NKV = 8, HD = 128;
  const int M = BS * SEQ;  // 4096 token rows

  char* ws = (char*)d_ws;
  const size_t MB = 1024 * 1024;
  unsigned short* xb  = (unsigned short*)(ws + 0);        // 32 MB
  unsigned short* wqb = (unsigned short*)(ws + 32 * MB);  // 32 MB
  unsigned short* wkb = (unsigned short*)(ws + 64 * MB);  //  8 MB
  unsigned short* wvb = (unsigned short*)(ws + 72 * MB);  //  8 MB
  unsigned short* wob = (unsigned short*)(ws + 80 * MB);  // 32 MB
  float*          qf  = (float*)(ws + 112 * MB);          // 64 MB (f32 Q pre-rope)
  float*          kf  = (float*)(ws + 176 * MB);          // 16 MB
  float*          vf  = (float*)(ws + 192 * MB);          // 16 MB
  unsigned short* qbb = (unsigned short*)(ws + 208 * MB); // 32 MB [b,32,s,128]
  unsigned short* kbb = (unsigned short*)(ws + 240 * MB); //  8 MB [b,8,s,128]
  unsigned short* vbt = (unsigned short*)(ws + 248 * MB); //  8 MB [b,8,128,s]
  unsigned short* obb = (unsigned short*)(ws + 112 * MB); // 32 MB, aliases qf (dead)

  // 1) convert inputs/weights to bf16
  f32_to_bf16_kernel<<<2048, 256, 0, stream>>>(x,  xb,  (size_t)M * D);
  f32_to_bf16_kernel<<<2048, 256, 0, stream>>>(wq, wqb, (size_t)D * D);
  f32_to_bf16_kernel<<<1024, 256, 0, stream>>>(wk, wkb, (size_t)NKV * HD * D);
  f32_to_bf16_kernel<<<1024, 256, 0, stream>>>(wv, wvb, (size_t)NKV * HD * D);
  f32_to_bf16_kernel<<<2048, 256, 0, stream>>>(wo, wob, (size_t)D * D);

  // 2) QKV projections (WMMA GEMMs, f32 accumulate)
  dim3 blk(256);
  gemm_bf16_nt_kernel<<<dim3(D / 128, M / 256), blk, 0, stream>>>(xb, wqb, qf, M, D, D);
  gemm_bf16_nt_kernel<<<dim3((NKV * HD) / 128, M / 256), blk, 0, stream>>>(xb, wkb, kf, M, NKV * HD, D);
  gemm_bf16_nt_kernel<<<dim3((NKV * HD) / 128, M / 256), blk, 0, stream>>>(xb, wvb, vf, M, NKV * HD, D);

  // 3) RoPE + pack to attention layouts (V transposed for PV B-fragments)
  rope_pack_kernel<<<4096, 256, 0, stream>>>(qf, freqs, qbb, NH, 0);
  rope_pack_kernel<<<2048, 256, 0, stream>>>(kf, freqs, kbb, NKV, 0);
  rope_pack_kernel<<<2048, 256, 0, stream>>>(vf, freqs, vbt, NKV, 1);

  // 4) causal flash attention (WMMA QK^T + PV)
  attn_kernel<<<dim3(SEQ / 128, BS * NH), blk, 0, stream>>>(qbb, kbb, vbt, obb);

  // 5) output projection -> f32 d_out
  gemm_bf16_nt_kernel<<<dim3(D / 128, M / 256), blk, 0, stream>>>(obb, wob, out, M, D, D);
}
